// Original_Net_73632919322712
// MI455X (gfx1250) — matrix operand order repaired and checked by reading, not compiled, not run on hardware
//
#include <hip/hip_runtime.h>
#include <hip/hip_bf16.h>
#include <stdint.h>

// Problem sizes (fixed by the reference)
#define B_SZ 4096
#define T_SZ 256
#define H_SZ 512

// Tiling
#define BM       32          // batch rows per workgroup
#define THREADS  512         // 16 waves (wave32)
#define NKT      16          // H / 32 k-tiles
#define HB_STRIDE 520        // h row stride in LDS elements (1040 B, bank-conflict pad)
#define WJ_STRIDE 40         // W slice per-j stride in LDS elements (80 B, bank-conflict pad)

typedef __attribute__((ext_vector_type(16))) __bf16 v16bf;
typedef __attribute__((ext_vector_type(8)))  __bf16 v8bf;
typedef __attribute__((ext_vector_type(8)))  float  v8f;

// GCC-style int4 to exactly match the async-to-LDS builtin's parameter type
typedef int v4i __attribute__((vector_size(16)));
typedef __attribute__((address_space(1))) v4i* gptr_v4i;
typedef __attribute__((address_space(3))) v4i* lptr_v4i;

#if __has_builtin(__builtin_amdgcn_s_wait_asynccnt)
#define WAIT_ASYNC0() __builtin_amdgcn_s_wait_asynccnt(0)
#else
#define WAIT_ASYNC0() asm volatile("s_wait_asynccnt 0" ::: "memory")
#endif

#if __has_builtin(__builtin_amdgcn_global_load_async_to_lds_b128)
#define HAVE_ASYNC_LDS 1
#else
#define HAVE_ASYNC_LDS 0
#endif

__device__ __forceinline__ v16bf cat8(v8bf a, v8bf b) {
  return __builtin_shufflevector(a, b, 0,1,2,3,4,5,6,7,8,9,10,11,12,13,14,15);
}

// Copy one 32 KB k-slice of W2 (k-slice-contiguous bf16 W_hh) into LDS.
// All 512 threads participate: lane j=tid copies its 64-byte row (4 x b128)
// into the padded LDS layout (stride 80 B).
__device__ __forceinline__ void copy_slice(const __bf16* __restrict__ W2,
                                           __bf16* dst, int slice, int tid) {
  const __bf16* g = W2 + ((size_t)slice * H_SZ + tid) * 32;
  __bf16* l = dst + tid * WJ_STRIDE;
#if HAVE_ASYNC_LDS
#pragma unroll
  for (int c = 0; c < 4; ++c) {
    __builtin_amdgcn_global_load_async_to_lds_b128(
        (gptr_v4i)(void*)(g + c * 8),
        (lptr_v4i)(void*)(l + c * 8),
        0, 0);
  }
#else
#pragma unroll
  for (int c = 0; c < 4; ++c)
    *(uint4*)(void*)(l + c * 8) = *(const uint4*)(const void*)(g + c * 8);
#endif
}

// Prep: W2[(kt*512 + j)*32 + kk] = bf16(W_hh[j*512 + kt*32 + kk])
// (B-matrix wants column j of W_hh^T == row j of W_hh, k-contiguous per slice)
__global__ void prep_w2_kernel(const float* __restrict__ Whh,
                               __bf16* __restrict__ W2) {
  int i = blockIdx.x * blockDim.x + threadIdx.x;   // 0 .. H*H-1
  int kk  = i & 31;
  int tmp = i >> 5;
  int j   = tmp & (H_SZ - 1);
  int kt  = tmp >> 9;
  W2[i] = (__bf16)Whh[j * H_SZ + kt * 32 + kk];
}

__global__ __launch_bounds__(THREADS, 1)
void rnn_scan_kernel(const float* __restrict__ x,     // (B,T)
                     const float* __restrict__ w_hi,  // (H)
                     const float* __restrict__ b_hi,  // (H)
                     const __bf16* __restrict__ W2,   // k-sliced bf16 W_hh
                     const float* __restrict__ W_yh,  // (1,H)
                     const float* __restrict__ b_yh,  // (1)
                     const float* __restrict__ h0,    // (1,H)
                     float* __restrict__ out)         // (B,1)
{
  __shared__ __align__(16) __bf16 hbuf[BM * HB_STRIDE];
  __shared__ __align__(16) __bf16 wbuf[2][H_SZ * WJ_STRIDE];

  const int tid      = threadIdx.x;
  const int lane     = tid & 31;
  const int wave     = tid >> 5;
  const int rowblock = wave & 1;    // 2 row blocks of 16
  const int colgroup = wave >> 1;   // 8 col groups of 64 (4 N-tiles)
  const int hi       = lane >> 4;   // K-half selector
  const int lo       = lane & 15;   // row (A) / column (B,C)
  const int wg_row0  = blockIdx.x * BM;

  // init h = broadcast(h0) in bf16
  for (int idx = tid; idx < BM * H_SZ; idx += THREADS) {
    int r = idx >> 9;
    int c = idx & (H_SZ - 1);
    hbuf[r * HB_STRIDE + c] = (__bf16)h0[c];
  }

  // per-lane loop-invariant bias/input weights for this wave's 4 N-tiles
  float whi_r[4], bhi_r[4];
#pragma unroll
  for (int nt = 0; nt < 4; ++nt) {
    int col = colgroup * 64 + nt * 16 + lo;
    whi_r[nt] = w_hi[col];
    bhi_r[nt] = b_hi[col];
  }

  // prefetch slice 0 into buffer 0
  copy_slice(W2, wbuf[0], 0, tid);
  WAIT_ASYNC0();
  __syncthreads();

  v8f acc[4];
#pragma unroll
  for (int nt = 0; nt < 4; ++nt)
    acc[nt] = (v8f){0.f,0.f,0.f,0.f,0.f,0.f,0.f,0.f};

  for (int t = 0; t < T_SZ; ++t) {
    for (int kt = 0; kt < NKT; ++kt) {
      // prefetch next W slice into the other buffer (overlaps WMMA below)
      copy_slice(W2, wbuf[(kt + 1) & 1], (kt + 1) & (NKT - 1), tid);

      // A fragment: row = lo, K halves at kt*32 + hi*8 and +16 (ISA layout)
      const __bf16* arow =
          &hbuf[(rowblock * 16 + lo) * HB_STRIDE + kt * 32 + hi * 8];
      v8bf a0 = *(const v8bf*)(const void*)(arow);
      v8bf a1 = *(const v8bf*)(const void*)(arow + 16);
      v16bf a = cat8(a0, a1);

      const __bf16* wb = wbuf[kt & 1];
#pragma unroll
      for (int nt = 0; nt < 4; ++nt) {
        // B fragment: column j, K halves at hi*8 and +16 (matches A)
        int j = colgroup * 64 + nt * 16 + lo;
        const __bf16* bp = &wb[j * WJ_STRIDE + hi * 8];
        v8bf b0 = *(const v8bf*)(const void*)(bp);
        v8bf b1 = *(const v8bf*)(const void*)(bp + 16);
        v16bf b = cat8(b0, b1);
        acc[nt] = __builtin_amdgcn_wmma_f32_16x16x32_bf16(
            false, a, false, b, (short)0, acc[nt], false, false);
      }
      WAIT_ASYNC0();
      __syncthreads();
    }

    // epilogue: h_new = relu(acc + x_t*w_hi + b_hi), write back to LDS (bf16)
    float xv[8];
#pragma unroll
    for (int v = 0; v < 8; ++v) {
      int rg = wg_row0 + rowblock * 16 + v + 8 * hi;  // C layout: M = v + 8*hi
      xv[v] = x[(size_t)rg * T_SZ + t];
    }
#pragma unroll
    for (int nt = 0; nt < 4; ++nt) {
      int c = colgroup * 64 + nt * 16 + lo;
#pragma unroll
      for (int v = 0; v < 8; ++v) {
        float val = acc[nt][v] + xv[v] * whi_r[nt] + bhi_r[nt];
        val = fmaxf(val, 0.0f);
        int r = rowblock * 16 + v + 8 * hi;
        hbuf[r * HB_STRIDE + c] = (__bf16)val;
      }
      acc[nt] = (v8f){0.f,0.f,0.f,0.f,0.f,0.f,0.f,0.f};
    }
    __syncthreads();
  }

  // y = h_final @ W_yh^T + b_yh  (O = 1): one thread per batch row
  if (tid < BM) {
    float accy = b_yh[0];
    for (int k = 0; k < H_SZ; ++k)
      accy += (float)hbuf[tid * HB_STRIDE + k] * W_yh[k];
    out[wg_row0 + tid] = accy;
  }
}

extern "C" void kernel_launch(void* const* d_in, const int* in_sizes, int n_in,
                              void* d_out, int out_size, void* d_ws, size_t ws_size,
                              hipStream_t stream) {
  const float* x    = (const float*)d_in[0];  // (B,T)
  const float* W_hi = (const float*)d_in[1];  // (H,1)
  const float* b_hi = (const float*)d_in[2];  // (H)
  const float* W_hh = (const float*)d_in[3];  // (H,H)
  const float* W_yh = (const float*)d_in[4];  // (1,H)
  const float* b_yh = (const float*)d_in[5];  // (1)
  const float* h0   = (const float*)d_in[6];  // (1,H)
  float* out = (float*)d_out;

  __bf16* W2 = (__bf16*)d_ws;                 // H*H bf16 = 512 KB scratch

  prep_w2_kernel<<<(H_SZ * H_SZ) / 256, 256, 0, stream>>>(W_hh, W2);
  rnn_scan_kernel<<<B_SZ / BM, THREADS, 0, stream>>>(
      x, W_hi, b_hi, W2, W_yh, b_yh, h0, out);
}
